// ReductionCell_71871982731581
// MI455X (gfx1250) — matrix operand order repaired and checked by reading, not compiled, not run on hardware
//
#include <hip/hip_runtime.h>

// ---------------- problem constants ----------------
#define B_     32
#define IMGSZ  224
#define CIN_   3
#define ED_    64          // embed dim per branch / conv channels / TD
#define HP_    56          // output spatial (224/4)
#define TT_    3136        // 56*56 tokens
#define H1_    112
#define T1_    12544       // 112*112
#define DIM_   256         // 4*ED
#define NKQV_  192
#define MPF_   32          // performer features

typedef _Float16 h8   __attribute__((ext_vector_type(8)));
typedef _Float16 v16h __attribute__((ext_vector_type(16)));
typedef float    v8f  __attribute__((ext_vector_type(8)));

// ---- WMMA fragment helpers (CDNA5 16x16x32 f16, wave32) ----
// A (16x32, 16-bit) per ISA 7.12.2: lane m=lane&15, kh=lane>>4;
//   elements 0..7  -> k = kbase + 8*kh + e
//   elements 8..15 -> k = kbase + 16 + 8*kh + (e-8)
// B (32x16): lane n=lane&15, kh=lane>>4; same k order as A (two 8-half chunks)
//   (weights stored [N][K], K padded to x32)
__device__ inline v16h afrag(h8 lo, h8 hi) {
  v16h a;
#pragma unroll
  for (int i = 0; i < 8; ++i) { a[i] = lo[i]; a[i + 8] = hi[i]; }
  return a;
}
__device__ inline v16h bfrag(const _Float16* __restrict__ w, int n, int K, int kbase, int kh) {
  const _Float16* p = w + (size_t)n * K + kbase + 8 * kh;
  return afrag(*(const h8*)p, *(const h8*)(p + 16));
}
__device__ inline v8f wmma32(v16h a, v16h b, v8f c) {
  return __builtin_amdgcn_wmma_f32_16x16x32_f16(false, a, false, b, (short)0, c, false, false);
}
// branchless zero-select of an h8 (lowers to 4x v_cndmask on b32 halves)
__device__ inline h8 sel8(bool ok, h8 v) {
  union { h8 h; int i[4]; } u;
  u.h = v;
#pragma unroll
  for (int i = 0; i < 4; ++i) u.i[i] = ok ? u.i[i] : 0;
  return u.h;
}
__device__ inline int clampi(int v, int lo, int hi) { return v < lo ? lo : (v > hi ? hi : v); }
__device__ inline float gelu_f(float v) { return 0.5f * v * (1.f + erff(v * 0.70710678118654752f)); }
__device__ inline float silu_f(float v) { return v / (1.f + expf(-v)); }

// ---------------- weight prep (fp32 -> f16, [N][K] with padding) ----------------
__global__ void k_prep_prm(const float* __restrict__ s, _Float16* __restrict__ d) {
  const int i = blockIdx.x * 256 + threadIdx.x;              // 4*64*160
  if (i >= 4 * 64 * 160) return;
  const int k = i % 160, dn = i / 160;                       // k = c*49+ky*7+kx
  d[i] = (k < 147) ? (_Float16)s[dn * 147 + k] : (_Float16)0.f;
}
__global__ void k_prep_pcm1(const float* __restrict__ s, _Float16* __restrict__ d) {
  const int i = blockIdx.x * 256 + threadIdx.x;              // 64*32
  if (i >= 64 * 32) return;
  const int k = i % 32, n = i / 32;                          // k = c*9+ky*3+kx
  d[i] = (k < 27) ? (_Float16)s[n * 27 + k] : (_Float16)0.f;
}
__global__ void k_prep_pcm23(const float* __restrict__ s, _Float16* __restrict__ d) {
  const int i = blockIdx.x * 256 + threadIdx.x;              // 64*576
  if (i >= 64 * 576) return;
  const int k = i % 576, n = i / 576;                        // k = tap*64 + c  (tap=ky*3+kx)
  const int tap = k >> 6, c = k & 63;
  const int ky = tap / 3, kx = tap - ky * 3;
  d[i] = (_Float16)s[((n * 64 + c) * 3 + ky) * 3 + kx];
}
__global__ void k_prep_f16(const float* __restrict__ s, _Float16* __restrict__ d, int n) {
  const int i = blockIdx.x * 256 + threadIdx.x;
  if (i < n) d[i] = (_Float16)s[i];
}
__global__ void k_prep_bn(const float* __restrict__ g, const float* __restrict__ be,
                          const float* __restrict__ mean, const float* __restrict__ var,
                          const float* __restrict__ cb, float* __restrict__ scale,
                          float* __restrict__ shift) {
  const int i = threadIdx.x;
  if (i < 64) {
    const float inv = g[i] * rsqrtf(var[i] + 1e-5f);
    scale[i] = inv;
    shift[i] = (cb[i] - mean[i]) * inv + be[i];
  }
}

// ---------------- PRM: 4 dilated 7x7 s4 convs + GELU (implicit GEMM, K=147->160) ----------------
__global__ void k_conv_prm(const float* __restrict__ x, const _Float16* __restrict__ w,
                           const float* __restrict__ bias, float* __restrict__ prm_x) {
  const int lane = threadIdx.x & 31, wv = threadIdx.x >> 5;
  const int d = blockIdx.z, b = blockIdx.y;
  const int t0 = (blockIdx.x * 4 + wv) * 16;
  const int dil = d + 1, pad = 3 * dil - 1;                  // ceil((6d-3)/2)
  const _Float16* wb = w + d * 64 * 160;
  const int m = lane & 15, kh = lane >> 4;
  const int t = t0 + m, oy = t / HP_, ox = t - oy * HP_;
  const int oy4 = oy * 4 - pad, ox4 = ox * 4 - pad;
  const float* xb = x + (size_t)b * CIN_ * IMGSZ * IMGSZ;

  v8f acc[4] = {};
#pragma unroll
  for (int ks = 0; ks < 5; ++ks) {
    const int kbase = ks * 32;
    v16h a;
#pragma unroll
    for (int e = 0; e < 16; ++e) {
      const int k = kbase + ((e < 8) ? (8 * kh + e) : (8 + 8 * kh + e));
      const int c = k / 49, rem = k - c * 49;
      const int ky = rem / 7, kx = rem - ky * 7;
      const int iy = oy4 + ky * dil;
      const int ix = ox4 + kx * dil;
      const bool ok = (k < 147) & ((unsigned)iy < IMGSZ) & ((unsigned)ix < IMGSZ);
      // clamp to a legal address, load unconditionally, cndmask to zero
      const int cc = clampi(c, 0, CIN_ - 1);
      const int iyc = clampi(iy, 0, IMGSZ - 1);
      const int ixc = clampi(ix, 0, IMGSZ - 1);
      const float v = xb[(size_t)(cc * IMGSZ + iyc) * IMGSZ + ixc];
      a[e] = ok ? (_Float16)v : (_Float16)0.f;
    }
#pragma unroll
    for (int nt = 0; nt < 4; ++nt)
      acc[nt] = wmma32(a, bfrag(wb, nt * 16 + m, 160, kbase, kh), acc[nt]);
  }
  const float* bp = bias + d * 64;
  float* ob = prm_x + (size_t)b * TT_ * DIM_ + d * 64;
#pragma unroll
  for (int nt = 0; nt < 4; ++nt) {
    const int n = nt * 16 + m;
#pragma unroll
    for (int r = 0; r < 8; ++r) {
      const int row = t0 + r + 8 * kh;
      ob[(size_t)row * DIM_ + n] = gelu_f(acc[nt][r] + bp[n]);
    }
  }
}

// ---------------- pcm1: 3->64 3x3 s2 + SiLU, out NHWC f16 (112x112) ----------------
__global__ void k_conv_pcm1(const float* __restrict__ x, const _Float16* __restrict__ w,
                            const float* __restrict__ bias, _Float16* __restrict__ c1) {
  const int lane = threadIdx.x & 31, wv = threadIdx.x >> 5;
  const int b = blockIdx.y;
  const int t0 = (blockIdx.x * 4 + wv) * 16;
  const int m = lane & 15, kh = lane >> 4;
  const int t = t0 + m, oy = t / H1_, ox = t - oy * H1_;
  const int oy2 = oy * 2 - 1, ox2 = ox * 2 - 1;
  const float* xb = x + (size_t)b * CIN_ * IMGSZ * IMGSZ;

  v16h a;
#pragma unroll
  for (int e = 0; e < 16; ++e) {
    const int k = (e < 8) ? (8 * kh + e) : (8 + 8 * kh + e);
    const int c = k / 9, rem = k - c * 9;
    const int ky = rem / 3, kx = rem - ky * 3;
    const int iy = oy2 + ky, ix = ox2 + kx;
    const bool ok = (k < 27) & ((unsigned)iy < IMGSZ) & ((unsigned)ix < IMGSZ);
    const int cc = clampi(c, 0, CIN_ - 1);
    const int iyc = clampi(iy, 0, IMGSZ - 1);
    const int ixc = clampi(ix, 0, IMGSZ - 1);
    const float v = xb[(size_t)(cc * IMGSZ + iyc) * IMGSZ + ixc];
    a[e] = ok ? (_Float16)v : (_Float16)0.f;
  }
  v8f acc[4] = {};
#pragma unroll
  for (int nt = 0; nt < 4; ++nt)
    acc[nt] = wmma32(a, bfrag(w, nt * 16 + m, 32, 0, kh), acc[nt]);
  _Float16* ob = c1 + (size_t)b * T1_ * ED_;
#pragma unroll
  for (int nt = 0; nt < 4; ++nt) {
    const int n = nt * 16 + m;
#pragma unroll
    for (int r = 0; r < 8; ++r) {
      const int row = t0 + r + 8 * kh;
      ob[(size_t)row * ED_ + n] = (_Float16)silu_f(acc[nt][r] + bias[n]);
    }
  }
}

// ---------------- pcm2: 64->64 3x3 s2 + BN + SiLU (K=576, NHWC f16 in/out) ----------------
__global__ void k_conv_pcm2(const _Float16* __restrict__ c1, const _Float16* __restrict__ w,
                            const float* __restrict__ scale, const float* __restrict__ shift,
                            _Float16* __restrict__ c2) {
  const int lane = threadIdx.x & 31, wv = threadIdx.x >> 5;
  const int b = blockIdx.y;
  const int t0 = (blockIdx.x * 4 + wv) * 16;
  const int m = lane & 15, kh = lane >> 4;
  const int t = t0 + m, oy = t / HP_, ox = t - oy * HP_;
  const int oy2 = oy * 2 - 1, ox2 = ox * 2 - 1;
  const _Float16* ib = c1 + (size_t)b * T1_ * ED_;

  v8f acc[4] = {};
#pragma unroll
  for (int ks = 0; ks < 18; ++ks) {
    const int kbase = ks * 32;
    const int k0 = kbase + 8 * kh;               // both 8-chunks land in the same 64-wide tap
    const int tap = k0 >> 6, c0 = k0 & 63;
    const int ky = tap / 3, kx = tap - ky * 3;
    const int iy = oy2 + ky, ix = ox2 + kx;
    const bool ok = ((unsigned)iy < H1_) & ((unsigned)ix < H1_);
    const int iyc = clampi(iy, 0, H1_ - 1);
    const int ixc = clampi(ix, 0, H1_ - 1);
    const _Float16* p = ib + (size_t)(iyc * H1_ + ixc) * ED_ + c0;
    const h8 lo = sel8(ok, *(const h8*)p);
    const h8 hi = sel8(ok, *(const h8*)(p + 16));
    const v16h a = afrag(lo, hi);
#pragma unroll
    for (int nt = 0; nt < 4; ++nt)
      acc[nt] = wmma32(a, bfrag(w, nt * 16 + m, 576, kbase, kh), acc[nt]);
  }
  _Float16* ob = c2 + (size_t)b * TT_ * ED_;
#pragma unroll
  for (int nt = 0; nt < 4; ++nt) {
    const int n = nt * 16 + m;
#pragma unroll
    for (int r = 0; r < 8; ++r) {
      const int row = t0 + r + 8 * kh;
      ob[(size_t)row * ED_ + n] = (_Float16)silu_f(acc[nt][r] * scale[n] + shift[n]);
    }
  }
}

// ---------------- pcm3: 64->64 3x3 s1 + bias + SiLU -> convX f32 ----------------
__global__ void k_conv_pcm3(const _Float16* __restrict__ c2, const _Float16* __restrict__ w,
                            const float* __restrict__ bias, float* __restrict__ convX) {
  const int lane = threadIdx.x & 31, wv = threadIdx.x >> 5;
  const int b = blockIdx.y;
  const int t0 = (blockIdx.x * 4 + wv) * 16;
  const int m = lane & 15, kh = lane >> 4;
  const int t = t0 + m, oy = t / HP_, ox = t - oy * HP_;
  const _Float16* ib = c2 + (size_t)b * TT_ * ED_;

  v8f acc[4] = {};
#pragma unroll
  for (int ks = 0; ks < 18; ++ks) {
    const int kbase = ks * 32;
    const int k0 = kbase + 8 * kh;
    const int tap = k0 >> 6, c0 = k0 & 63;
    const int ky = tap / 3, kx = tap - ky * 3;
    const int iy = oy - 1 + ky, ix = ox - 1 + kx;
    const bool ok = ((unsigned)iy < HP_) & ((unsigned)ix < HP_);
    const int iyc = clampi(iy, 0, HP_ - 1);
    const int ixc = clampi(ix, 0, HP_ - 1);
    const _Float16* p = ib + (size_t)(iyc * HP_ + ixc) * ED_ + c0;
    const h8 lo = sel8(ok, *(const h8*)p);
    const h8 hi = sel8(ok, *(const h8*)(p + 16));
    const v16h a = afrag(lo, hi);
#pragma unroll
    for (int nt = 0; nt < 4; ++nt)
      acc[nt] = wmma32(a, bfrag(w, nt * 16 + m, 576, kbase, kh), acc[nt]);
  }
  float* ob = convX + (size_t)b * TT_ * ED_;
#pragma unroll
  for (int nt = 0; nt < 4; ++nt) {
    const int n = nt * 16 + m;
#pragma unroll
    for (int r = 0; r < 8; ++r) {
      const int row = t0 + r + 8 * kh;
      ob[(size_t)row * ED_ + n] = silu_f(acc[nt][r] + bias[n]);
    }
  }
}

// ---------------- LN over 256 (wave32 per row) -> xn f16 ----------------
__global__ void k_ln1(const float* __restrict__ xin, const float* __restrict__ g,
                      const float* __restrict__ be, _Float16* __restrict__ xn) {
  const int lane = threadIdx.x & 31, wv = threadIdx.x >> 5;
  const size_t row = (size_t)blockIdx.x * 8 + wv;
  const float* p = xin + row * DIM_;
  float v[8]; float s = 0.f;
#pragma unroll
  for (int i = 0; i < 8; ++i) { v[i] = p[i * 32 + lane]; s += v[i]; }
#pragma unroll
  for (int o = 16; o > 0; o >>= 1) s += __shfl_xor(s, o, 32);
  const float mu = s * (1.f / 256.f);
  float s2 = 0.f;
#pragma unroll
  for (int i = 0; i < 8; ++i) { const float dd = v[i] - mu; s2 += dd * dd; }
#pragma unroll
  for (int o = 16; o > 0; o >>= 1) s2 += __shfl_xor(s2, o, 32);
  const float rstd = rsqrtf(s2 * (1.f / 256.f) + 1e-5f);
  _Float16* ob = xn + row * DIM_;
#pragma unroll
  for (int i = 0; i < 8; ++i) {
    const int j = i * 32 + lane;
    ob[j] = (_Float16)((v[i] - mu) * rstd * g[j] + be[j]);
  }
}

// ---------------- KQV GEMM: (rows x 256) @ (256 x 192) + bias -> f32 ----------------
__global__ void k_gemm_kqv(const _Float16* __restrict__ xn, const _Float16* __restrict__ w,
                           const float* __restrict__ bias, float* __restrict__ out) {
  const int lane = threadIdx.x & 31, wv = threadIdx.x >> 5;
  const int t0 = (blockIdx.x * 4 + wv) * 16;
  const int m = lane & 15, kh = lane >> 4;
  const _Float16* ar = xn + (size_t)(t0 + m) * DIM_;

  v8f acc[12] = {};
#pragma unroll
  for (int ks = 0; ks < 8; ++ks) {
    const int kbase = ks * 32;
    const v16h a = afrag(*(const h8*)(ar + kbase + 8 * kh),
                         *(const h8*)(ar + kbase + 16 + 8 * kh));
#pragma unroll
    for (int nt = 0; nt < 12; ++nt)
      acc[nt] = wmma32(a, bfrag(w, nt * 16 + m, 256, kbase, kh), acc[nt]);
  }
#pragma unroll
  for (int nt = 0; nt < 12; ++nt) {
    const int n = nt * 16 + m;
#pragma unroll
    for (int r = 0; r < 8; ++r) {
      const size_t row = t0 + r + 8 * kh;
      out[row * NKQV_ + n] = acc[nt][r] + bias[n];
    }
  }
}

// ---------------- Performer features: kp=exp(Wk - ||k||^2/2)/sqrt(M) etc ----------------
__global__ void k_perf(const float* __restrict__ kqv, const float* __restrict__ wperf,
                       float* __restrict__ kp, float* __restrict__ qpf,
                       _Float16* __restrict__ qph) {
  const int lane = threadIdx.x & 31, wv = threadIdx.x >> 5;
  const size_t row = (size_t)blockIdx.x * 8 + wv;
  const float* kk = kqv + row * NKQV_;
  const float* qq = kk + 64;
  const float* wm = wperf + lane * 64;                       // lane == feature m
  float wk = 0.f, wq = 0.f, xk = 0.f, xq = 0.f;
#pragma unroll 8
  for (int i = 0; i < 64; ++i) {
    const float kv = kk[i], qv = qq[i], wv_ = wm[i];
    wk += kv * wv_; wq += qv * wv_;
    xk += kv * kv;  xq += qv * qv;
  }
  const float inv = 0.17677669529663689f;                    // 1/sqrt(32)
  const float kpv = expf(wk - 0.5f * xk) * inv;
  const float qpv = expf(wq - 0.5f * xq) * inv;
  kp[row * MPF_ + lane] = kpv;
  qpf[row * MPF_ + lane] = qpv;
  qph[row * MPF_ + lane] = (_Float16)qpv;
}

// ---------------- kptv[b][n][m] = sum_t v*kp ; kpsum[b][m] = sum_t kp ----------------
__global__ void k_kptv(const float* __restrict__ kqv, const float* __restrict__ kp,
                       _Float16* __restrict__ kptvh, float* __restrict__ kpsum) {
  __shared__ float sv[32 * 64];
  __shared__ float sk[32 * 32];
  const int b = blockIdx.x, tid = threadIdx.x;
  const int n = tid >> 2, m0 = (tid & 3) * 8;
  float acc[8] = {0.f};
  float ksm = 0.f;
  for (int t0 = 0; t0 < TT_; t0 += 32) {
    for (int idx = tid; idx < 2048; idx += 256) {
      const int tt = idx >> 6, nn = idx & 63;
      sv[idx] = kqv[((size_t)b * TT_ + t0 + tt) * NKQV_ + 128 + nn];
    }
    for (int idx = tid; idx < 1024; idx += 256) {
      const int tt = idx >> 5, mm = idx & 31;
      sk[idx] = kp[((size_t)b * TT_ + t0 + tt) * MPF_ + mm];
    }
    __syncthreads();
    for (int tt = 0; tt < 32; ++tt) {
      const float vv = sv[tt * 64 + n];
#pragma unroll
      for (int j = 0; j < 8; ++j) acc[j] += vv * sk[tt * 32 + m0 + j];
    }
    if (tid < 32)
      for (int tt = 0; tt < 32; ++tt) ksm += sk[tt * 32 + tid];
    __syncthreads();
  }
#pragma unroll
  for (int j = 0; j < 8; ++j)
    kptvh[((size_t)b * 64 + n) * 32 + m0 + j] = (_Float16)acc[j];
  if (tid < 32) kpsum[b * 32 + tid] = ksm;
}

// ---------------- attention out: xo = v + (qp@kptv^T)/D @ projW^T + proj_b + convX ----------------
__global__ void k_attn(const float* __restrict__ qpf, const _Float16* __restrict__ qph,
                       const _Float16* __restrict__ kptvh, const float* __restrict__ kpsum,
                       const float* __restrict__ kqv, const float* __restrict__ convX,
                       const _Float16* __restrict__ wproj, const float* __restrict__ projb,
                       float* __restrict__ xo) {
  __shared__ alignas(32) _Float16 tile[4][16][64];
  const int lane = threadIdx.x & 31, wv = threadIdx.x >> 5;
  const int t0 = (blockIdx.x * 4 + wv) * 16;
  const int b = t0 / TT_;                                    // tiles never cross batch (3136 % 16 == 0)
  const int m = lane & 15, kh = lane >> 4;

  // D = qp . sum_t kp  (one row per lane 0..15)
  float invD = 0.f;
  if (kh == 0) {
    const float* qr = qpf + (size_t)(t0 + m) * MPF_;
    const float* ks = kpsum + b * MPF_;
    float D = 1e-8f;
#pragma unroll
    for (int i = 0; i < 32; ++i) D += qr[i] * ks[i];
    invD = 1.f / D;
  }

  // ya0 = qp @ kptv^T  (K=32, one WMMA step)
  const _Float16* ar = qph + (size_t)(t0 + m) * MPF_;
  const v16h a = afrag(*(const h8*)(ar + 8 * kh), *(const h8*)(ar + 16 + 8 * kh));
  const _Float16* kb = kptvh + (size_t)b * 64 * 32;
  v8f ya[4];
#pragma unroll
  for (int nt = 0; nt < 4; ++nt) {
    v8f z = {};
    ya[nt] = wmma32(a, bfrag(kb, nt * 16 + m, 32, 0, kh), z);
  }
  // scale by 1/D, stash as f16 A-tile in LDS
#pragma unroll
  for (int nt = 0; nt < 4; ++nt) {
#pragma unroll
    for (int r = 0; r < 8; ++r) {
      const float di = __shfl(invD, r + 8 * kh, 32);
      tile[wv][r + 8 * kh][nt * 16 + m] = (_Float16)(ya[nt][r] * di);
    }
  }
  __syncthreads();

  // proj: C = v + convX + proj_b ; accumulate ya0 @ projW^T (K=64)
  v8f acc[4];
#pragma unroll
  for (int nt = 0; nt < 4; ++nt) {
    const int n = nt * 16 + m;
#pragma unroll
    for (int r = 0; r < 8; ++r) {
      const size_t row = t0 + r + 8 * kh;
      acc[nt][r] = kqv[row * NKQV_ + 128 + n] + convX[row * ED_ + n] + projb[n];
    }
  }
  const _Float16* tr = &tile[wv][m][0];
#pragma unroll
  for (int ks2 = 0; ks2 < 2; ++ks2) {
    const int kbase = ks2 * 32;
    const v16h a2 = afrag(*(const h8*)(tr + kbase + 8 * kh),
                          *(const h8*)(tr + kbase + 16 + 8 * kh));
#pragma unroll
    for (int nt = 0; nt < 4; ++nt)
      acc[nt] = wmma32(a2, bfrag(wproj, nt * 16 + m, 64, kbase, kh), acc[nt]);
  }
#pragma unroll
  for (int nt = 0; nt < 4; ++nt) {
    const int n = nt * 16 + m;
#pragma unroll
    for (int r = 0; r < 8; ++r) {
      const size_t row = t0 + r + 8 * kh;
      xo[row * ED_ + n] = acc[nt][r];
    }
  }
}

// ---------------- LN over 64 -> xn2 f16 ----------------
__global__ void k_ln2(const float* __restrict__ xo, const float* __restrict__ g,
                      const float* __restrict__ be, _Float16* __restrict__ xn2) {
  const int lane = threadIdx.x & 31, wv = threadIdx.x >> 5;
  const size_t row = (size_t)blockIdx.x * 8 + wv;
  const float* p = xo + row * ED_;
  const float v0 = p[lane], v1 = p[lane + 32];
  float s = v0 + v1;
#pragma unroll
  for (int o = 16; o > 0; o >>= 1) s += __shfl_xor(s, o, 32);
  const float mu = s * (1.f / 64.f);
  const float d0 = v0 - mu, d1 = v1 - mu;
  float s2 = d0 * d0 + d1 * d1;
#pragma unroll
  for (int o = 16; o > 0; o >>= 1) s2 += __shfl_xor(s2, o, 32);
  const float rstd = rsqrtf(s2 * (1.f / 64.f) + 1e-5f);
  _Float16* ob = xn2 + row * ED_;
  ob[lane]      = (_Float16)(d0 * rstd * g[lane] + be[lane]);
  ob[lane + 32] = (_Float16)(d1 * rstd * g[lane + 32] + be[lane + 32]);
}

// ---------------- MLP: out = xo + gelu(xn2@W1^T+b1)@W2^T + b2 ----------------
__global__ void k_mlp(const _Float16* __restrict__ xn2, const _Float16* __restrict__ w1,
                      const float* __restrict__ b1, const _Float16* __restrict__ w2,
                      const float* __restrict__ b2, const float* __restrict__ xo,
                      float* __restrict__ out) {
  __shared__ alignas(32) _Float16 tile[4][16][64];
  const int lane = threadIdx.x & 31, wv = threadIdx.x >> 5;
  const int t0 = (blockIdx.x * 4 + wv) * 16;
  const int m = lane & 15, kh = lane >> 4;

  v8f acc[4];
#pragma unroll
  for (int nt = 0; nt < 4; ++nt) {
    const float bv = b1[nt * 16 + m];
#pragma unroll
    for (int r = 0; r < 8; ++r) acc[nt][r] = bv;
  }
  const _Float16* ar = xn2 + (size_t)(t0 + m) * ED_;
#pragma unroll
  for (int ks = 0; ks < 2; ++ks) {
    const int kbase = ks * 32;
    const v16h a = afrag(*(const h8*)(ar + kbase + 8 * kh),
                         *(const h8*)(ar + kbase + 16 + 8 * kh));
#pragma unroll
    for (int nt = 0; nt < 4; ++nt)
      acc[nt] = wmma32(a, bfrag(w1, nt * 16 + m, 64, kbase, kh), acc[nt]);
  }
#pragma unroll
  for (int nt = 0; nt < 4; ++nt)
#pragma unroll
    for (int r = 0; r < 8; ++r)
      tile[wv][r + 8 * kh][nt * 16 + m] = (_Float16)gelu_f(acc[nt][r]);
  __syncthreads();

  v8f acc2[4];
#pragma unroll
  for (int nt = 0; nt < 4; ++nt) {
    const int n = nt * 16 + m;
#pragma unroll
    for (int r = 0; r < 8; ++r) {
      const size_t row = t0 + r + 8 * kh;
      acc2[nt][r] = xo[row * ED_ + n] + b2[n];
    }
  }
  const _Float16* tr = &tile[wv][m][0];
#pragma unroll
  for (int ks = 0; ks < 2; ++ks) {
    const int kbase = ks * 32;
    const v16h a2 = afrag(*(const h8*)(tr + kbase + 8 * kh),
                          *(const h8*)(tr + kbase + 16 + 8 * kh));
#pragma unroll
    for (int nt = 0; nt < 4; ++nt)
      acc2[nt] = wmma32(a2, bfrag(w2, nt * 16 + m, 64, kbase, kh), acc2[nt]);
  }
#pragma unroll
  for (int nt = 0; nt < 4; ++nt) {
    const int n = nt * 16 + m;
#pragma unroll
    for (int r = 0; r < 8; ++r) {
      const size_t row = t0 + r + 8 * kh;
      out[row * ED_ + n] = acc2[nt][r];
    }
  }
}

// ---------------- host ----------------
extern "C" void kernel_launch(void* const* d_in, const int* in_sizes, int n_in,
                              void* d_out, int out_size, void* d_ws, size_t ws_size,
                              hipStream_t stream) {
  (void)in_sizes; (void)n_in; (void)out_size; (void)ws_size;
  const float* x       = (const float*)d_in[0];
  const float* prm_w   = (const float*)d_in[1];
  const float* prm_b   = (const float*)d_in[2];
  const float* pcm_w1  = (const float*)d_in[3];
  const float* pcm_b1  = (const float*)d_in[4];
  const float* pcm_w2  = (const float*)d_in[5];
  const float* pcm_b2  = (const float*)d_in[6];
  const float* bn_g    = (const float*)d_in[7];
  const float* bn_b    = (const float*)d_in[8];
  const float* bn_mean = (const float*)d_in[9];
  const float* bn_var  = (const float*)d_in[10];
  const float* pcm_w3  = (const float*)d_in[11];
  const float* pcm_b3  = (const float*)d_in[12];
  const float* ln1_g   = (const float*)d_in[13];
  const float* ln1_b   = (const float*)d_in[14];
  const float* kqv_w   = (const float*)d_in[15];
  const float* kqv_b   = (const float*)d_in[16];
  const float* w_perf  = (const float*)d_in[17];
  const float* proj_w  = (const float*)d_in[18];
  const float* proj_b  = (const float*)d_in[19];
  const float* ln2_g   = (const float*)d_in[20];
  const float* ln2_b   = (const float*)d_in[21];
  const float* mlp_w1  = (const float*)d_in[22];
  const float* mlp_b1  = (const float*)d_in[23];
  const float* mlp_w2  = (const float*)d_in[24];
  const float* mlp_b2  = (const float*)d_in[25];

  size_t off = 0;
  char* ws = (char*)d_ws;
  auto alloc = [&](size_t bytes) -> void* {
    void* p = ws + off;
    off += (bytes + 255) & ~(size_t)255;
    return p;
  };
  const size_t ROWS = (size_t)B_ * TT_;                      // 100352
  _Float16* wprm    = (_Float16*)alloc(4 * 64 * 160 * 2);
  _Float16* wpcm1   = (_Float16*)alloc(64 * 32 * 2);
  _Float16* wpcm2   = (_Float16*)alloc(64 * 576 * 2);
  _Float16* wpcm3   = (_Float16*)alloc(64 * 576 * 2);
  _Float16* wkqv    = (_Float16*)alloc(192 * 256 * 2);
  _Float16* wproj   = (_Float16*)alloc(64 * 64 * 2);
  _Float16* wmlp1   = (_Float16*)alloc(64 * 64 * 2);
  _Float16* wmlp2   = (_Float16*)alloc(64 * 64 * 2);
  float*    bnscale = (float*)alloc(64 * 4);
  float*    bnshift = (float*)alloc(64 * 4);
  float*    prmx    = (float*)alloc(ROWS * DIM_ * 4);        // 102.8 MB
  _Float16* xnbuf   = (_Float16*)alloc(ROWS * DIM_ * 2);     //  51.4 MB
  _Float16* c1buf   = (_Float16*)alloc((size_t)B_ * T1_ * ED_ * 2); // 51.4 MB NHWC
  _Float16* c2buf   = (_Float16*)alloc((size_t)B_ * TT_ * ED_ * 2); // 12.8 MB NHWC
  float*    convXb  = (float*)alloc(ROWS * ED_ * 4);         //  25.7 MB
  float*    kqvbuf  = (float*)alloc(ROWS * NKQV_ * 4);       //  77.1 MB
  float*    kpbuf   = (float*)alloc(ROWS * MPF_ * 4);        //  12.8 MB
  float*    qpf     = (float*)alloc(ROWS * MPF_ * 4);        //  12.8 MB
  _Float16* qph     = (_Float16*)alloc(ROWS * MPF_ * 2);     //   6.4 MB
  _Float16* kptvh   = (_Float16*)alloc((size_t)B_ * 64 * 32 * 2);
  float*    kpsum   = (float*)alloc((size_t)B_ * 32 * 4);
  float*    xobuf   = (float*)alloc(ROWS * ED_ * 4);         //  25.7 MB
  _Float16* xn2buf  = (_Float16*)alloc(ROWS * ED_ * 2);      //  12.8 MB

  // weight prep
  k_prep_prm  <<<160, 256, 0, stream>>>(prm_w, wprm);
  k_prep_pcm1 <<<8,   256, 0, stream>>>(pcm_w1, wpcm1);
  k_prep_pcm23<<<144, 256, 0, stream>>>(pcm_w2, wpcm2);
  k_prep_pcm23<<<144, 256, 0, stream>>>(pcm_w3, wpcm3);
  k_prep_f16  <<<192, 256, 0, stream>>>(kqv_w, wkqv, 192 * 256);
  k_prep_f16  <<<16,  256, 0, stream>>>(proj_w, wproj, 64 * 64);
  k_prep_f16  <<<16,  256, 0, stream>>>(mlp_w1, wmlp1, 64 * 64);
  k_prep_f16  <<<16,  256, 0, stream>>>(mlp_w2, wmlp2, 64 * 64);
  k_prep_bn   <<<1,   64,  0, stream>>>(bn_g, bn_b, bn_mean, bn_var, pcm_b2, bnscale, bnshift);

  // main pipeline
  k_conv_prm <<<dim3(49, 32, 4), 128, 0, stream>>>(x, wprm, prm_b, prmx);
  k_conv_pcm1<<<dim3(196, 32),   128, 0, stream>>>(x, wpcm1, pcm_b1, c1buf);
  k_conv_pcm2<<<dim3(49, 32),    128, 0, stream>>>(c1buf, wpcm2, bnscale, bnshift, c2buf);
  k_conv_pcm3<<<dim3(49, 32),    128, 0, stream>>>(c2buf, wpcm3, pcm_b3, convXb);
  k_ln1      <<<12544, 256, 0, stream>>>(prmx, ln1_g, ln1_b, xnbuf);
  k_gemm_kqv <<<1568,  128, 0, stream>>>(xnbuf, wkqv, kqv_b, kqvbuf);
  k_perf     <<<12544, 256, 0, stream>>>(kqvbuf, w_perf, kpbuf, qpf, qph);
  k_kptv     <<<32,    256, 0, stream>>>(kqvbuf, kpbuf, kptvh, kpsum);
  k_attn     <<<1568,  128, 0, stream>>>(qpf, qph, kptvh, kpsum, kqvbuf, convXb, wproj, proj_b, xobuf);
  k_ln2      <<<12544, 256, 0, stream>>>(xobuf, ln2_g, ln2_b, xn2buf);
  k_mlp      <<<1568,  128, 0, stream>>>(xn2buf, wmlp1, mlp_b1, wmlp2, mlp_b2, xobuf, (float*)d_out);
}
